// GCMCModel_50302656971283
// MI455X (gfx1250) — matrix operand order repaired and checked by reading, not compiled, not run on hardware
//
#include <hip/hip_runtime.h>
#include <hip/hip_bf16.h>
#include <stdint.h>

typedef __attribute__((ext_vector_type(16))) _Float16 v16h;
typedef __attribute__((ext_vector_type(8)))  _Float16 v8h;
typedef __attribute__((ext_vector_type(8)))  float    v8f;
typedef __attribute__((ext_vector_type(4)))  unsigned int v4u;
typedef __attribute__((ext_vector_type(8)))  unsigned int v8u;
typedef __attribute__((ext_vector_type(8)))  int      v8i;
typedef __attribute__((ext_vector_type(4)))  int      v4i;

#if defined(__has_builtin)
#  if __has_builtin(__builtin_amdgcn_tensor_load_to_lds)
#    define HAVE_TDM 1
#  endif
#endif
#ifndef HAVE_TDM
#  define HAVE_TDM 0
#endif

// Definitive compile-time probe: if this warning does NOT appear in stderr,
// the device pass compiled the real TDM path.
#if !HAVE_TDM && defined(__AMDGCN__)
#  warning "CDNA5 TDM builtin unavailable on device pass - using fallback copy"
#endif

#define KCHUNK 160   // int32 elements per TDM tile column-chunk (= 5 WMMA K-steps)
#define NCHUNK 125   // 125 * 160 = 20000

// ---------------------------------------------------------------------------
// Helpers
// ---------------------------------------------------------------------------

// 16-bit A-matrix 16x32 layout (contiguous, matching B): lane<16 holds K = 0..15,
// lane>=16 holds K = 16..31; half index h -> K offset:
__device__ __forceinline__ int klist(int h, bool hi) {
    return h + (hi ? 16 : 0);
}

// Load 16 contiguous f16 values (32B) as a v16h (two b128 loads).
__device__ __forceinline__ v16h load16h(const _Float16* p) {
    const v8h* q = (const v8h*)p;
    v8h lo = q[0];
    v8h hi = q[1];
    v16h r;
#pragma unroll
    for (int i = 0; i < 8; ++i) { r[i] = lo[i]; r[i + 8] = hi[i]; }
    return r;
}

__device__ __forceinline__ v8f wmma_f16(v16h a, v16h b, v8f c) {
    // D = A(16x32 f16) * B(32x16 f16) + C(16x16 f32)
    return __builtin_amdgcn_wmma_f32_16x16x32_f16(
        /*neg_a=*/false, a, /*neg_b=*/false, b,
        /*c_mod=*/(short)0, c, /*reuse_a=*/false, /*reuse_b=*/false);
}

// Pack two 0/1 int32 edge values into one dword of two f16 (0.0 / 1.0):
// (e0 | e1<<16) * 0x3C00 ; exact, no cross-half carry since e*0x3C00 < 2^16.
__device__ __forceinline__ unsigned int pack01(int e0, int e1) {
    return ((unsigned int)e0 | ((unsigned int)e1 << 16)) * 0x3C00u;
}

#if HAVE_TDM
// Issue one TDM 2D tile load: 16 rows x KCHUNK int32, global row stride 20000
// elements, into LDS at byte offset lds_off. Tracked by TENSORcnt.
__device__ __forceinline__ void tdm_load_tile(const int* gptr, unsigned int lds_off) {
    unsigned long long ga = (unsigned long long)(uintptr_t)gptr;
    v4u g0;
    g0[0] = 1u;                                   // count=1, user descriptor
    g0[1] = lds_off;                              // lds_addr (bytes)
    g0[2] = (unsigned int)(ga & 0xFFFFFFFFu);     // global_addr[31:0]
    g0[3] = (unsigned int)((ga >> 32) & 0x1FFFFFFu) | (2u << 30); // addr[56:32], type=2
    v8i g1;
    g1[0] = (int)(2u << 16);                      // data_size = 4B; no multicast/pad
    g1[1] = (int)(20000u << 16);                  // tensor_dim0[15:0] in bits[31:16]
    g1[2] = (int)((20000u >> 16) | (5120u << 16));// tensor_dim0 hi / tensor_dim1 lo
    g1[3] = (int)((5120u >> 16) | ((unsigned)KCHUNK << 16)); // dim1 hi / tile_dim0
    g1[4] = 16;                                   // tile_dim1 = 16 rows, tile_dim2 = 0
    g1[5] = 20000;                                // tensor_dim0_stride (elements)
    g1[6] = 0;
    g1[7] = 0;
    v4i z4 = {0, 0, 0, 0};
#if __clang_major__ >= 23
    v8i z8 = {0, 0, 0, 0, 0, 0, 0, 0};
    __builtin_amdgcn_tensor_load_to_lds(g0, g1, z4, z4, z8, 0);
#else
    __builtin_amdgcn_tensor_load_to_lds(g0, g1, z4, z4, 0);
#endif
}
#endif

// ---------------------------------------------------------------------------
// Prep: transpose-convert f32 (R x C) -> f16 (C x R)
// ---------------------------------------------------------------------------
__global__ void cvtT_kernel(const float* __restrict__ src, _Float16* __restrict__ dst,
                            int R, int C) {
    int i = blockIdx.x * blockDim.x + threadIdx.x;
    int total = R * C;
    if (i < total) {
        int r = i / C;
        int c = i % C;
        dst[(size_t)c * R + r] = (_Float16)src[i];
    }
}

// ---------------------------------------------------------------------------
// Heavy kernel: agg = edge(16xK 0/1) @ emb(Kx32), then fused GCN
//   gcn = relu(agg @ W + b), stored into gcn_h (1024 x 160) at cols r*32..
// grid: (64 tiles, 5 ratings, 2 sides), block: 64 (2 waves)
// Edge tiles are streamed with the Tensor Data Mover into per-wave
// double-buffered LDS tiles; waves overlap TDM with pack+WMMA.
// ---------------------------------------------------------------------------
__global__ __launch_bounds__(64)
void agg_gcn_kernel(const int* __restrict__ eIU, const int* __restrict__ eUI,
                    const _Float16* __restrict__ embT_u, const _Float16* __restrict__ embT_i,
                    const _Float16* __restrict__ wgT_u,  const _Float16* __restrict__ wgT_i,
                    const float* __restrict__ gb_u,      const float* __restrict__ gb_i,
                    float* __restrict__ gh_u,            float* __restrict__ gh_i) {
    const int tile = blockIdx.x;   // 0..63 (16 batch rows each)
    const int r    = blockIdx.y;   // 0..4
    const int side = blockIdx.z;   // 0 = user path (eIU,U), 1 = item path (eUI,I)
    const int tid  = threadIdx.x;
    const int lane = tid & 31;
    const int wave = tid >> 5;     // 0..1
    const bool hi  = lane >= 16;
    const int  m   = lane & 15;

    const int*      edge = side ? eUI    : eIU;
    const _Float16* embT = side ? embT_i : embT_u;
    const _Float16* wgT  = side ? wgT_i  : wgT_u;
    const float*    gb   = side ? gb_i   : gb_u;
    float*          gh   = side ? gh_i   : gh_u;

    // Per-wave double-buffered edge tile: [wave][buf][16 rows x KCHUNK]
    __shared__ __align__(16) int ebuf[2][2][16 * KCHUNK];   // 40 KB
    __shared__ float aggLDS[16][32];                        //  2 KB

    const int* gtile = edge + ((size_t)r * 1024 + (size_t)tile * 16) * 20000;
    const _Float16* bcol0 = embT + (size_t)m * 20000;          // emb column m
    const _Float16* bcol1 = embT + (size_t)(m + 16) * 20000;   // emb column m+16

    v8f c0 = {};  // cols 0..15
    v8f c1 = {};  // cols 16..31

#if HAVE_TDM
    const unsigned int ldsA = (unsigned int)(uintptr_t)&ebuf[wave][0][0];
    const unsigned int ldsB = (unsigned int)(uintptr_t)&ebuf[wave][1][0];
    tdm_load_tile(gtile + (size_t)wave * KCHUNK, ldsA);   // prime the pipeline
#endif

    int cur = 0;
    for (int s = wave; s < NCHUNK; s += 2) {
        const int nxt = s + 2;
#if HAVE_TDM
        if (nxt < NCHUNK) {
            tdm_load_tile(gtile + (size_t)nxt * KCHUNK, cur ? ldsA : ldsB);
            __builtin_amdgcn_s_wait_tensorcnt(1);   // oldest (current) tile landed
        } else {
            __builtin_amdgcn_s_wait_tensorcnt(0);
        }
#else
        {   // cooperative fallback: wave copies its own tile (b128 both sides)
            int* dst = &ebuf[wave][cur][0];
            for (int i = lane * 4; i < 16 * KCHUNK; i += 128) {
                int rr = i / KCHUNK, cc = i % KCHUNK;
                *(int4*)&dst[i] =
                    *(const int4*)(gtile + (size_t)rr * 20000 + (size_t)s * KCHUNK + cc);
            }
        }
#endif
        const int* lt = &ebuf[wave][cur][0];
        const int kbase = s * KCHUNK;
#pragma unroll
        for (int kk = 0; kk < 5; ++kk) {
            // A fragment from LDS tile: 16 x {0,1} int32 -> packed f16 0.0/1.0
            const int* lrow = lt + m * KCHUNK + kk * 32 + (hi ? 16 : 0);
            int4 q0 = *(const int4*)(lrow);
            int4 q1 = *(const int4*)(lrow + 4);
            int4 q2 = *(const int4*)(lrow + 8);
            int4 q3 = *(const int4*)(lrow + 12);
            union { v8u u; v16h h; } au;
            au.u[0] = pack01(q0.x, q0.y);
            au.u[1] = pack01(q0.z, q0.w);
            au.u[2] = pack01(q1.x, q1.y);
            au.u[3] = pack01(q1.z, q1.w);
            au.u[4] = pack01(q2.x, q2.y);
            au.u[5] = pack01(q2.z, q2.w);
            au.u[6] = pack01(q3.x, q3.y);
            au.u[7] = pack01(q3.z, q3.w);
            v16h a = au.h;

            const int kb = kbase + kk * 32 + (hi ? 16 : 0);
            v16h b0 = load16h(bcol0 + kb);
            v16h b1 = load16h(bcol1 + kb);

            c0 = wmma_f16(a, b0, c0);
            c1 = wmma_f16(a, b1, c1);
        }
        cur ^= 1;
    }

    // Cross-wave reduction of the 16x32 partial agg in LDS
    for (int i = tid; i < 512; i += 64) ((float*)aggLDS)[i] = 0.0f;
    __syncthreads();
    {
        const int mrb = hi ? 8 : 0;
#pragma unroll
        for (int v = 0; v < 8; ++v) {
            atomicAdd(&aggLDS[v + mrb][m],      c0[v]);
            atomicAdd(&aggLDS[v + mrb][m + 16], c1[v]);
        }
    }
    __syncthreads();

    // Fused GCN: relu(agg @ W + b). All waves compute (keeps EXEC all-ones
    // for WMMA); wave 0 stores.
    v16h a;
#pragma unroll
    for (int h = 0; h < 16; ++h) a[h] = (_Float16)aggLDS[m][klist(h, hi)];
    const int kb2 = hi ? 16 : 0;
    v16h b0 = load16h(wgT + (size_t)m * 32 + kb2);
    v16h b1 = load16h(wgT + (size_t)(m + 16) * 32 + kb2);
    v8f z = {};
    v8f d0 = wmma_f16(a, b0, z);
    v8f d1 = wmma_f16(a, b1, z);

    if (tid < 32) {
        const int mrb = hi ? 8 : 0;
#pragma unroll
        for (int v = 0; v < 8; ++v) {
            const size_t row = (size_t)tile * 16 + v + mrb;
            gh[row * 160 + r * 32 + m]      = fmaxf(d0[v] + gb[m],      0.0f);
            gh[row * 160 + r * 32 + m + 16] = fmaxf(d1[v] + gb[m + 16], 0.0f);
        }
    }
}

// ---------------------------------------------------------------------------
// Head kernel: fc1(user/item) -> feature products -> l1 -> l2 -> l3 + biases
// grid: 64 tiles, block: 32 (one wave)
// ---------------------------------------------------------------------------
__global__ __launch_bounds__(32)
void head_kernel(const int* __restrict__ x,
                 const float* __restrict__ user_emb, const float* __restrict__ item_emb,
                 const float* __restrict__ gh_u,     const float* __restrict__ gh_i,
                 const _Float16* __restrict__ wfT_u, const _Float16* __restrict__ wfT_i,
                 const float* __restrict__ fb_u,     const float* __restrict__ fb_i,
                 const _Float16* __restrict__ wl1T,  const float* __restrict__ l1b,
                 const _Float16* __restrict__ wl2T,  const float* __restrict__ l2b,
                 const float* __restrict__ l3w,      const float* __restrict__ l3b,
                 const float* __restrict__ ubias,    const float* __restrict__ ibias,
                 float* __restrict__ out) {
    const int tile = blockIdx.x;
    const int lane = threadIdx.x & 31;
    const bool hi  = lane >= 16;
    const int  m   = lane & 15;
    const int  mrb = hi ? 8 : 0;

    __shared__ int   uid[16], iid[16];
    __shared__ float ue[16][32], ie[16][32], guo[16][32], gio[16][32];
    __shared__ float h1s[16][64], h2s[16][32];

    if (lane < 16) {
        uid[lane] = x[(tile * 16 + lane) * 2];
        iid[lane] = x[(tile * 16 + lane) * 2 + 1];
    }
    __syncthreads();

    // Gather user/item embedding rows for this tile
    for (int idx = lane; idx < 512; idx += 32) {
        int rw = idx >> 5, e = idx & 31;
        ue[rw][e] = user_emb[(size_t)uid[rw] * 32 + e];
        ie[rw][e] = item_emb[(size_t)iid[rw] * 32 + e];
    }
    __syncthreads();

    // fc1: (16 x 160) @ (160 x 32) + b, for user and item sides
    for (int side = 0; side < 2; ++side) {
        const float*    gh = side ? gh_i  : gh_u;
        const _Float16* wf = side ? wfT_i : wfT_u;
        const float*    fb = side ? fb_i  : fb_u;
        const float* grow = gh + ((size_t)tile * 16 + m) * 160;
        v8f d0 = {}, d1 = {};
        for (int kc = 0; kc < 5; ++kc) {
            v16h a;
#pragma unroll
            for (int h = 0; h < 16; ++h) a[h] = (_Float16)grow[kc * 32 + klist(h, hi)];
            const int kb = kc * 32 + (hi ? 16 : 0);
            v16h b0 = load16h(wf + (size_t)m * 160 + kb);
            v16h b1 = load16h(wf + (size_t)(m + 16) * 160 + kb);
            d0 = wmma_f16(a, b0, d0);
            d1 = wmma_f16(a, b1, d1);
        }
        float (*dst)[32] = side ? gio : guo;
#pragma unroll
        for (int v = 0; v < 8; ++v) {
            dst[v + mrb][m]      = d0[v] + fb[m];
            dst[v + mrb][m + 16] = d1[v] + fb[m + 16];
        }
    }
    __syncthreads();

    // l1: feat(16 x 128) @ (128 x 64) + b, relu.
    // feat K-chunk kc == concat block: 0: ue*ie, 1: ue*gio, 2: guo*ie, 3: guo*gio
    v8f acc[4] = {{}, {}, {}, {}};
    for (int kc = 0; kc < 4; ++kc) {
        float (*X)[32] = (kc < 2) ? ue  : guo;
        float (*Y)[32] = (kc & 1) ? gio : ie;
        v16h a;
#pragma unroll
        for (int h = 0; h < 16; ++h) {
            int kk = klist(h, hi);
            a[h] = (_Float16)(X[m][kk] * Y[m][kk]);
        }
        const int kb = kc * 32 + (hi ? 16 : 0);
#pragma unroll
        for (int nh = 0; nh < 4; ++nh) {
            v16h b = load16h(wl1T + (size_t)(nh * 16 + m) * 128 + kb);
            acc[nh] = wmma_f16(a, b, acc[nh]);
        }
    }
#pragma unroll
    for (int nh = 0; nh < 4; ++nh)
#pragma unroll
        for (int v = 0; v < 8; ++v)
            h1s[v + mrb][nh * 16 + m] = fmaxf(acc[nh][v] + l1b[nh * 16 + m], 0.0f);
    __syncthreads();

    // l2: (16 x 64) @ (64 x 32) + b, relu
    v8f e0 = {}, e1 = {};
    for (int kc = 0; kc < 2; ++kc) {
        v16h a;
#pragma unroll
        for (int h = 0; h < 16; ++h) a[h] = (_Float16)h1s[m][kc * 32 + klist(h, hi)];
        const int kb = kc * 32 + (hi ? 16 : 0);
        v16h b0 = load16h(wl2T + (size_t)m * 64 + kb);
        v16h b1 = load16h(wl2T + (size_t)(m + 16) * 64 + kb);
        e0 = wmma_f16(a, b0, e0);
        e1 = wmma_f16(a, b1, e1);
    }
#pragma unroll
    for (int v = 0; v < 8; ++v) {
        h2s[v + mrb][m]      = fmaxf(e0[v] + l2b[m],      0.0f);
        h2s[v + mrb][m + 16] = fmaxf(e1[v] + l2b[m + 16], 0.0f);
    }
    __syncthreads();

    // l3: (16 x 32) @ (32 x 1) + b + user_bias + item_bias
    if (lane < 16) {
        float s = l3b[0];
#pragma unroll
        for (int e = 0; e < 32; ++e) s += h2s[lane][e] * l3w[e];
        out[tile * 16 + lane] = s + ubias[uid[lane]] + ibias[iid[lane]];
    }
}

// ---------------------------------------------------------------------------
// Launcher
// ---------------------------------------------------------------------------
extern "C" void kernel_launch(void* const* d_in, const int* in_sizes, int n_in,
                              void* d_out, int out_size, void* d_ws, size_t ws_size,
                              hipStream_t stream) {
    const int*   x        = (const int*)d_in[0];
    const int*   edge_UI  = (const int*)d_in[1];
    const int*   edge_IU  = (const int*)d_in[2];
    const float* user_e   = (const float*)d_in[3];
    const float* item_e   = (const float*)d_in[4];
    const float* gcn_u_w  = (const float*)d_in[5];
    const float* gcn_u_b  = (const float*)d_in[6];
    const float* gcn_i_w  = (const float*)d_in[7];
    const float* gcn_i_b  = (const float*)d_in[8];
    const float* fc1_u_w  = (const float*)d_in[9];
    const float* fc1_u_b  = (const float*)d_in[10];
    const float* fc1_i_w  = (const float*)d_in[11];
    const float* fc1_i_b  = (const float*)d_in[12];
    const float* l1_w     = (const float*)d_in[13];
    const float* l1_b     = (const float*)d_in[14];
    const float* l2_w     = (const float*)d_in[15];
    const float* l2_b     = (const float*)d_in[16];
    const float* l3_w     = (const float*)d_in[17];
    const float* l3_b     = (const float*)d_in[18];
    const float* u_bias   = (const float*)d_in[19];
    const float* i_bias   = (const float*)d_in[20];
    float* out = (float*)d_out;

    // Workspace carve-up (256B aligned regions)
    char* base = (char*)d_ws;
    size_t off = 0;
    auto carve = [&](size_t bytes) -> char* {
        char* p = base + off;
        off = (off + bytes + 255) & ~(size_t)255;
        return p;
    };
    _Float16* embT_u = (_Float16*)carve(20000 * 32 * sizeof(_Float16));
    _Float16* embT_i = (_Float16*)carve(20000 * 32 * sizeof(_Float16));
    _Float16* wgT_u  = (_Float16*)carve(32 * 32 * sizeof(_Float16));
    _Float16* wgT_i  = (_Float16*)carve(32 * 32 * sizeof(_Float16));
    _Float16* wfT_u  = (_Float16*)carve(160 * 32 * sizeof(_Float16));
    _Float16* wfT_i  = (_Float16*)carve(160 * 32 * sizeof(_Float16));
    _Float16* wl1T   = (_Float16*)carve(128 * 64 * sizeof(_Float16));
    _Float16* wl2T   = (_Float16*)carve(64 * 32 * sizeof(_Float16));
    float*    gh_u   = (float*)carve(1024 * 160 * sizeof(float));
    float*    gh_i   = (float*)carve(1024 * 160 * sizeof(float));
    (void)ws_size; (void)n_in; (void)in_sizes; (void)out_size;

    // Prep: transpose-convert embeddings and weights to f16 column-major
    auto cvt = [&](const float* src, _Float16* dst, int R, int C) {
        int total = R * C;
        cvtT_kernel<<<(total + 255) / 256, 256, 0, stream>>>(src, dst, R, C);
    };
    cvt(user_e,  embT_u, 20000, 32);
    cvt(item_e,  embT_i, 20000, 32);
    cvt(gcn_u_w, wgT_u,  32, 32);
    cvt(gcn_i_w, wgT_i,  32, 32);
    cvt(fc1_u_w, wfT_u,  160, 32);
    cvt(fc1_i_w, wfT_i,  160, 32);
    cvt(l1_w,    wl1T,   128, 64);
    cvt(l2_w,    wl2T,   64, 32);

    // Heavy aggregation + fused GCN (TDM-streamed edge tiles)
    agg_gcn_kernel<<<dim3(64, 5, 2), 64, 0, stream>>>(
        edge_IU, edge_UI, embT_u, embT_i, wgT_u, wgT_i, gcn_u_b, gcn_i_b, gh_u, gh_i);

    // Head: fc1 + feature products + MLP + biases
    head_kernel<<<64, 32, 0, stream>>>(
        x, user_e, item_e, gh_u, gh_i, wfT_u, wfT_i, fc1_u_b, fc1_i_b,
        wl1T, l1_b, wl2T, l2_b, l3_w, l3_b, u_bias, i_bias, out);
}
